// SimpleRNN_58042188038782
// MI455X (gfx1250) — matrix operand order repaired and checked by reading, not compiled, not run on hardware
//
#include <hip/hip_runtime.h>

// ---------------------------------------------------------------------------
// SimpleRNN fused kernel for MI455X (gfx1250, wave32, WMMA).
//   B=256, T=512, I=64, H=256, O=16
// Kernel 1 (rnn_scan): 16 workgroups (one per 16-row batch tile) x 16 waves
//   (one per 16-col N-tile of H). h kept in double-buffered LDS across all
//   512 timesteps; W_h2h / W_i2h held in registers as WMMA B-fragments.
//   Per step: 8 bf16 WMMAs (h @ W_h2h^T) + 2 bf16 WMMAs (x_t @ W_i2h^T, fused
//   input projection), then clamp(relu(pre),1) epilogue.
// Kernel 2 (out_proj): rnn_out @ W_out^T, N=16 == O, one WMMA tile per wave.
// ---------------------------------------------------------------------------

typedef __bf16 bf16_t;
typedef __attribute__((ext_vector_type(16))) __bf16 v16bf;
typedef __attribute__((ext_vector_type(8)))  float  v8f;

#define B_ 256
#define T_ 512
#define I_ 64
#define H_ 256
#define O_ 16

__device__ __forceinline__ bf16_t f2bf(float f) {
    unsigned u = __builtin_bit_cast(unsigned, f);
    unsigned r = (u + 0x7FFFu + ((u >> 16) & 1u)) >> 16;   // round-to-nearest-even
    unsigned short s = (unsigned short)r;
    return __builtin_bit_cast(bf16_t, s);
}
__device__ __forceinline__ float bf2f(bf16_t b) {
    unsigned short s = __builtin_bit_cast(unsigned short, b);
    unsigned u = ((unsigned)s) << 16;
    return __builtin_bit_cast(float, u);
}

// A-matrix 16x32 bf16 fragment from LDS, row-major [16][ldk] at column kOff.
// ISA layout: lane = M (0..15), half-wave selects K sub-blocks; K pairs packed.
__device__ __forceinline__ v16bf load_a_frag_lds(const bf16_t* lds, int ldk, int kOff) {
    const int lane = threadIdx.x & 31;
    const int m    = lane & 15;
    const int half = lane >> 4;
    const bf16_t* row = lds + m * ldk + kOff;
    v16bf a;
#pragma unroll
    for (int v = 0; v < 8; ++v) {
        const int kb = (v < 4) ? (2 * v + 8 * half) : (16 + 2 * (v - 4) + 8 * half);
        a[2 * v]     = row[kb];
        a[2 * v + 1] = row[kb + 1];
    }
    return a;
}

// B-matrix 32x16 bf16 fragment built from global f32 weight W[n][k] (row-major,
// stride ldk), taking B[k][n] = W[n0+n][kOff+k]. ISA layout: lane%16 = N,
// element e -> K = 8*(lane/16) + e (e<8), 16 + 8*(lane/16) + (e-8) (e>=8).
__device__ __forceinline__ v16bf load_b_frag_global(const float* __restrict__ W,
                                                    int ldk, int n0, int kOff) {
    const int lane = threadIdx.x & 31;
    const int n    = lane & 15;
    const int half = lane >> 4;
    const float* row = W + (size_t)(n0 + n) * ldk + kOff;
    v16bf b;
#pragma unroll
    for (int e = 0; e < 16; ++e) b[e] = f2bf(row[(e < 8) ? (8 * half + e) : (16 + 8 * half + (e - 8))]);
    return b;
}

__global__ __launch_bounds__(512)
void rnn_scan_kernel(const float* __restrict__ x,
                     const float* __restrict__ W_i2h, const float* __restrict__ b_i2h,
                     const float* __restrict__ W_h2h, const float* __restrict__ b_h2h,
                     const float* __restrict__ hidden_init,
                     float* __restrict__ rnn_out, float* __restrict__ hidden_out) {
    __shared__ bf16_t hbuf[2][16 * H_];   // double-buffered hidden tile, 16 KB
    __shared__ bf16_t xbuf[16 * I_];      // x_t tile, 2 KB
    __shared__ float  biasc[H_];          // b_i2h + b_h2h, 1 KB

    const int tid  = threadIdx.x;
    const int wave = tid >> 5;
    const int lane = tid & 31;
    const int n0   = wave * 16;           // N-tile of H owned by this wave
    const int b0   = blockIdx.x * 16;     // batch-row tile
    const int n    = lane & 15;
    const int half = lane >> 4;

    // Preload weight B-fragments into registers (once, amortized over T=512).
    v16bf Bh[H_ / 32];                    // 8 frags: W_h2h^T, K = 256
#pragma unroll
    for (int j = 0; j < H_ / 32; ++j) Bh[j] = load_b_frag_global(W_h2h, H_, n0, 32 * j);
    v16bf Bi[I_ / 32];                    // 2 frags: W_i2h^T, K = 64
#pragma unroll
    for (int j = 0; j < I_ / 32; ++j) Bi[j] = load_b_frag_global(W_i2h, I_, n0, 32 * j);

    // Combined bias and initial hidden state.
    for (int idx = tid; idx < H_; idx += 512) biasc[idx] = b_i2h[idx] + b_h2h[idx];
    for (int idx = tid; idx < 16 * H_; idx += 512)
        hbuf[0][idx] = f2bf(hidden_init[idx & (H_ - 1)]);

    int p = 0;
    for (int t = 0; t < T_; ++t) {
        // Stage x_t tile (16 x 64 f32 -> bf16) cooperatively.
        for (int idx = tid; idx < 16 * I_; idx += 512) {
            const int m = idx >> 6, k = idx & 63;
            xbuf[idx] = f2bf(x[(size_t)(b0 + m) * T_ * I_ + (size_t)t * I_ + k]);
        }
        __syncthreads();

        // Prefetch next timestep's x rows (global_prefetch_b8).
        if (t + 1 < T_ && lane == 0)
            __builtin_prefetch(&x[(size_t)(b0 + wave) * T_ * I_ + (size_t)(t + 1) * I_], 0, 1);

        v8f acc = {};
#pragma unroll
        for (int j = 0; j < H_ / 32; ++j) {   // h_prev @ W_h2h^T
            v16bf a = load_a_frag_lds(&hbuf[p][0], H_, 32 * j);
            acc = __builtin_amdgcn_wmma_f32_16x16x32_bf16(false, a, false, Bh[j],
                                                          (short)0, acc, false, false);
        }
#pragma unroll
        for (int j = 0; j < I_ / 32; ++j) {   // x_t @ W_i2h^T (fused projection)
            v16bf a = load_a_frag_lds(&xbuf[0], I_, 32 * j);
            acc = __builtin_amdgcn_wmma_f32_16x16x32_bf16(false, a, false, Bi[j],
                                                          (short)0, acc, false, false);
        }

        // Epilogue: pre = acc + h_prev + bias; a = min(relu(pre), 1).
#pragma unroll
        for (int r = 0; r < 8; ++r) {
            const int m = r + 8 * half;
            const float hprev = bf2f(hbuf[p][m * H_ + n0 + n]);
            const float pre   = acc[r] + hprev + biasc[n0 + n];
            const float av    = fminf(fmaxf(pre, 0.0f), 1.0f);
            rnn_out[(size_t)(b0 + m) * T_ * H_ + (size_t)t * H_ + n0 + n] = av;
            if (t == T_ - 1)
                hidden_out[(size_t)(b0 + m) * H_ + n0 + n] = av;
            hbuf[p ^ 1][m * H_ + n0 + n] = f2bf(av);
        }
        __syncthreads();
        p ^= 1;
    }
}

__global__ __launch_bounds__(256)
void out_proj_kernel(const float* __restrict__ rnn_out,
                     const float* __restrict__ W_out, const float* __restrict__ b_out,
                     float* __restrict__ out) {
    const int wave = threadIdx.x >> 5;
    const int lane = threadIdx.x & 31;
    const int m0   = (blockIdx.x * 8 + wave) * 16;   // row tile into [B*T, H]
    const int n    = lane & 15;
    const int half = lane >> 4;

    v16bf Bw[H_ / 32];                                // W_out^T B-frags, N=16==O
#pragma unroll
    for (int j = 0; j < H_ / 32; ++j) Bw[j] = load_b_frag_global(W_out, H_, 0, 32 * j);

    v8f acc = {};
#pragma unroll
    for (int j = 0; j < H_ / 32; ++j) {
        const int m = lane & 15;
        const float* row = rnn_out + (size_t)(m0 + m) * H_ + 32 * j;
        v16bf a;
#pragma unroll
        for (int v = 0; v < 8; ++v) {
            const int kb = (v < 4) ? (2 * v + 8 * half) : (16 + 2 * (v - 4) + 8 * half);
            a[2 * v]     = f2bf(row[kb]);
            a[2 * v + 1] = f2bf(row[kb + 1]);
        }
        acc = __builtin_amdgcn_wmma_f32_16x16x32_bf16(false, a, false, Bw[j],
                                                      (short)0, acc, false, false);
    }
    const float bias = b_out[n];
#pragma unroll
    for (int r = 0; r < 8; ++r) {
        const int m = r + 8 * half;
        out[(size_t)(m0 + m) * O_ + n] = acc[r] + bias;
    }
}

extern "C" void kernel_launch(void* const* d_in, const int* in_sizes, int n_in,
                              void* d_out, int out_size, void* d_ws, size_t ws_size,
                              hipStream_t stream) {
    const float* x      = (const float*)d_in[0];
    const float* W_i2h  = (const float*)d_in[1];
    const float* b_i2h  = (const float*)d_in[2];
    const float* W_h2h  = (const float*)d_in[3];
    const float* b_h2h  = (const float*)d_in[4];
    const float* W_out  = (const float*)d_in[5];
    const float* b_out  = (const float*)d_in[6];
    const float* h_init = (const float*)d_in[7];

    float* out    = (float*)d_out;                       // [B,T,O]
    float* hidden = out + (size_t)B_ * T_ * O_;          // [B,H]
    float* rnn    = hidden + (size_t)B_ * H_;            // [B,T,H]

    rnn_scan_kernel<<<B_ / 16, 512, 0, stream>>>(x, W_i2h, b_i2h, W_h2h, b_h2h,
                                                 h_init, rnn, hidden);
    out_proj_kernel<<<(B_ * T_) / (16 * 8), 256, 0, stream>>>(rnn, W_out, b_out, out);
}
